// HybridGATModel_89945205113097
// MI455X (gfx1250) — compile-verified, matrix operand order checked by reading, not run on hardware
//
#include <hip/hip_runtime.h>
#include <hip/hip_bf16.h>
#include <math.h>
#include <stdint.h>

// ---------------- problem constants (match reference) ----------------
#define NN        50000
#define E_RAW     800000
#define EDGES     (E_RAW + NN)     // with self loops appended
#define NEG_SLOPE 0.2f
#define EPS_F     1e-16f

typedef __bf16 bf16_t;
typedef __attribute__((ext_vector_type(16))) __bf16 v16bf;
typedef __attribute__((ext_vector_type(8)))  float  v8f;
typedef __attribute__((ext_vector_type(4)))  float  f4;

union FragBF { v16bf v; f4 q[2]; };

// round-to-nearest-even f32 -> bf16 (pure bit ops: no reliance on fp conversions)
__device__ inline bf16_t f2bf(float f) {
  union { float f; unsigned u; } c; c.f = f;
  unsigned u = c.u;
  unsigned r = u + 0x7FFFu + ((u >> 16) & 1u);
  unsigned short h = (unsigned short)(r >> 16);
  return __builtin_bit_cast(bf16_t, h);
}

__device__ inline void atomicMaxF(float* addr, float v) {
  if (v >= 0.0f) atomicMax((int*)addr, __float_as_int(v));
  else           atomicMin((unsigned int*)addr, __float_as_uint(v));
}

// ---------------- elementwise / conversion kernels ----------------
__global__ void k_f32_to_bf16(const float* __restrict__ in, bf16_t* __restrict__ out, size_t n) {
  size_t i = (size_t)blockIdx.x * blockDim.x + threadIdx.x;
  if (i < n) out[i] = f2bf(in[i]);
}

// W [K,N] row-major f32 -> Wt [N,K] row-major bf16
__global__ void k_transpose_bf16(const float* __restrict__ W, bf16_t* __restrict__ Wt, int K, int N) {
  int i = blockIdx.x * blockDim.x + threadIdx.x;
  if (i < K * N) {
    int k = i / N, n = i % N;
    Wt[(size_t)n * K + k] = f2bf(W[i]);
  }
}

__global__ void k_fill(float* __restrict__ p, float v, size_t n) {
  size_t i = (size_t)blockIdx.x * blockDim.x + threadIdx.x;
  if (i < n) p[i] = v;
}

// ---------------- WMMA bf16 GEMM: C[M,N] = A[M,K] * Bt[N,K]^T ----------------
// one wave computes one 16x16 tile; fragments loaded per the CDNA5 16-bit
// A-matrix VGPR layout: lane = 16*half + r, two contiguous 16B chunks per lane.
__global__ void k_gemm_bf16(const bf16_t* __restrict__ A, const bf16_t* __restrict__ Bt,
                            float* __restrict__ C, const float* __restrict__ bias,
                            int M, int N, int K, int relu) {
  int gt   = blockIdx.x * blockDim.x + threadIdx.x;
  int wave = gt >> 5;
  int lane = gt & 31;
  int ntN  = N >> 4;
  int tm   = wave / ntN;
  int tn   = wave - tm * ntN;
  if (tm * 16 >= M) return;

  int r    = lane & 15;
  int half = lane >> 4;

  const f4* arow = reinterpret_cast<const f4*>(A + (size_t)(tm * 16 + r) * K);
  const f4* brow = reinterpret_cast<const f4*>(Bt + (size_t)(tn * 16 + r) * K);

  v8f acc = {0.f, 0.f, 0.f, 0.f, 0.f, 0.f, 0.f, 0.f};

  for (int k0 = 0; k0 < K; k0 += 32) {
    FragBF a, b;
    int i0 = (k0 >> 3) + half;       // in units of 8 bf16 (16 bytes)
    a.q[0] = arow[i0];
    a.q[1] = arow[i0 + 2];
    b.q[0] = brow[i0];
    b.q[1] = brow[i0 + 2];
    acc = __builtin_amdgcn_wmma_f32_16x16x32_bf16(
        /*neg_a=*/false, a.v, /*neg_b=*/false, b.v,
        /*c_mod=*/(short)0, acc, /*reuse_a=*/false, /*reuse_b=*/false);
  }

  int n = tn * 16 + r;
  float bval = bias ? bias[n] : 0.0f;
#pragma unroll
  for (int v = 0; v < 8; ++v) {
    int m = tm * 16 + v + 8 * half;
    float val = acc[v] + bval;
    if (relu) val = fmaxf(val, 0.0f);
    C[(size_t)m * N + n] = val;
  }
}

// ---------------- attention coefficient dots: one wave per (node, head) ----------------
__global__ void k_att_reduce(const float* __restrict__ h, const float* __restrict__ as,
                             const float* __restrict__ ad, float* __restrict__ oS,
                             float* __restrict__ oD, int H, int C) {
  int gt   = blockIdx.x * blockDim.x + threadIdx.x;
  int wave = gt >> 5;
  int lane = gt & 31;
  if (wave >= NN * H) return;
  int node = wave / H, hh = wave - node * H;
  const float* row = h + ((size_t)node * H + hh) * C;
  float ss = 0.f, sd = 0.f;
  for (int c = lane; c < C; c += 32) {
    float v = row[c];
    ss += v * as[hh * C + c];
    sd += v * ad[hh * C + c];
  }
  for (int o = 16; o; o >>= 1) { ss += __shfl_xor(ss, o, 32); sd += __shfl_xor(sd, o, 32); }
  if (lane == 0) { oS[(size_t)node * H + hh] = ss; oD[(size_t)node * H + hh] = sd; }
}

// ---------------- edge passes ----------------
__global__ void k_edge_max(const int* __restrict__ src, const int* __restrict__ dst,
                           const float* __restrict__ aS, const float* __restrict__ aD,
                           float* __restrict__ m, int H) {
  int t = blockIdx.x * blockDim.x + threadIdx.x;
  if (t >= EDGES * H) return;
  int hh = t % H, e = t / H;
  int s = (e < E_RAW) ? src[e] : (e - E_RAW);
  int d = (e < E_RAW) ? dst[e] : (e - E_RAW);
  float v = aS[(size_t)s * H + hh] + aD[(size_t)d * H + hh];
  v = (v > 0.f) ? v : v * NEG_SLOPE;
  atomicMaxF(&m[(size_t)d * H + hh], v);
}

__global__ void k_edge_exp(const int* __restrict__ src, const int* __restrict__ dst,
                           const float* __restrict__ aS, const float* __restrict__ aD,
                           const float* __restrict__ m, float* __restrict__ z,
                           float* __restrict__ ew, int H) {
  int t = blockIdx.x * blockDim.x + threadIdx.x;
  if (t >= EDGES * H) return;
  int hh = t % H, e = t / H;
  int s = (e < E_RAW) ? src[e] : (e - E_RAW);
  int d = (e < E_RAW) ? dst[e] : (e - E_RAW);
  float v = aS[(size_t)s * H + hh] + aD[(size_t)d * H + hh];
  v = (v > 0.f) ? v : v * NEG_SLOPE;
  float x = __expf(v - m[(size_t)d * H + hh]);
  ew[(size_t)e * H + hh] = x;
  atomicAdd(&z[(size_t)d * H + hh], x);
}

// alpha-weighted scatter-add of messages; one thread per (edge, head, 4 channels)
__global__ void k_aggregate(const int* __restrict__ src, const int* __restrict__ dst,
                            const float* __restrict__ ew, const float* __restrict__ z,
                            const float* __restrict__ hfeat, float* __restrict__ agg,
                            int H, int C) {
  size_t t = (size_t)blockIdx.x * blockDim.x + threadIdx.x;
  int cg = C >> 2;
  size_t total = (size_t)EDGES * H * cg;
  if (t >= total) return;
  int q = (int)(t % cg);
  size_t u = t / cg;
  int hh = (int)(u % H);
  int e  = (int)(u / H);
  int s = (e < E_RAW) ? src[e] : (e - E_RAW);
  int d = (e < E_RAW) ? dst[e] : (e - E_RAW);
  float alpha = ew[(size_t)e * H + hh] / (z[(size_t)d * H + hh] + EPS_F);
  const f4 hv = *reinterpret_cast<const f4*>(hfeat + ((size_t)s * H + hh) * C + (q << 2));
  float* o = agg + ((size_t)d * H + hh) * C + (q << 2);
  atomicAdd(o + 0, hv.x * alpha);
  atomicAdd(o + 1, hv.y * alpha);
  atomicAdd(o + 2, hv.z * alpha);
  atomicAdd(o + 3, hv.w * alpha);
}

// out = ELU(agg + bias) -> bf16 (feeds next WMMA GEMM)
__global__ void k_elu_bias_bf16(const float* __restrict__ agg, const float* __restrict__ bias,
                                bf16_t* __restrict__ outb, int F, size_t n) {
  size_t i = (size_t)blockIdx.x * blockDim.x + threadIdx.x;
  if (i >= n) return;
  float v = agg[i] + bias[i % F];
  v = (v > 0.f) ? v : (__expf(v) - 1.0f);
  outb[i] = f2bf(v);
}

// final fc2: one wave per node, dot over C=64
__global__ void k_head(const float* __restrict__ h3, const float* __restrict__ w,
                       const float* __restrict__ b, float* __restrict__ out, int C) {
  int gt   = blockIdx.x * blockDim.x + threadIdx.x;
  int wave = gt >> 5;
  int lane = gt & 31;
  if (wave >= NN) return;
  const float* row = h3 + (size_t)wave * C;
  float s = 0.f;
  for (int c = lane; c < C; c += 32) s += row[c] * w[c];
  for (int o = 16; o; o >>= 1) s += __shfl_xor(s, o, 32);
  if (lane == 0) out[wave] = s + b[0];
}

// ---------------- host-side orchestration ----------------
static inline int blocksFor(size_t n, int bs) { return (int)((n + bs - 1) / bs); }

extern "C" void kernel_launch(void* const* d_in, const int* in_sizes, int n_in,
                              void* d_out, int out_size, void* d_ws, size_t ws_size,
                              hipStream_t stream) {
  (void)in_sizes; (void)n_in; (void)out_size; (void)ws_size;

  const float* x    = (const float*)d_in[0];
  const int*   ei   = (const int*)d_in[1];
  const int*   srcI = ei;             // edge_index[0,:]
  const int*   dstI = ei + E_RAW;     // edge_index[1,:]
  const float* W1   = (const float*)d_in[2];
  const float* as1  = (const float*)d_in[3];
  const float* ad1  = (const float*)d_in[4];
  const float* b1   = (const float*)d_in[5];
  const float* W2   = (const float*)d_in[6];
  const float* as2  = (const float*)d_in[7];
  const float* ad2  = (const float*)d_in[8];
  const float* b2   = (const float*)d_in[9];
  const float* fc1w = (const float*)d_in[10];
  const float* fc1b = (const float*)d_in[11];
  const float* fc2w = (const float*)d_in[12];
  const float* fc2b = (const float*)d_in[13];
  float* out = (float*)d_out;

  char* ws = (char*)d_ws;
  size_t off = 0;
  auto alloc = [&](size_t bytes) -> char* {
    off = (off + 255) & ~(size_t)255;
    char* p = ws + off;
    off += bytes;
    return p;
  };

  bf16_t* xbf  = (bf16_t*)alloc((size_t)NN * 128 * 2);
  bf16_t* w1t  = (bf16_t*)alloc((size_t)256 * 128 * 2);
  float*  h1   = (float*) alloc((size_t)NN * 256 * 4);
  float*  a1s  = (float*) alloc((size_t)NN * 2 * 4);
  float*  a1d  = (float*) alloc((size_t)NN * 2 * 4);
  float*  m1   = (float*) alloc((size_t)NN * 2 * 4);
  float*  z1   = (float*) alloc((size_t)NN * 2 * 4);
  float*  ew1  = (float*) alloc((size_t)EDGES * 2 * 4);
  float*  agg1 = (float*) alloc((size_t)NN * 256 * 4);
  bf16_t* h1b  = (bf16_t*)alloc((size_t)NN * 256 * 2);
  bf16_t* w2t  = (bf16_t*)alloc((size_t)128 * 256 * 2);
  float*  h2   = (float*) alloc((size_t)NN * 128 * 4);
  float*  a2s  = (float*) alloc((size_t)NN * 4);
  float*  a2d  = (float*) alloc((size_t)NN * 4);
  float*  m2   = (float*) alloc((size_t)NN * 4);
  float*  z2   = (float*) alloc((size_t)NN * 4);
  float*  ew2  = (float*) alloc((size_t)EDGES * 4);
  float*  agg2 = (float*) alloc((size_t)NN * 128 * 4);
  bf16_t* h2b  = (bf16_t*)alloc((size_t)NN * 128 * 2);
  bf16_t* fc1t = (bf16_t*)alloc((size_t)64 * 128 * 2);
  float*  h3   = (float*) alloc((size_t)NN * 64 * 4);

  const int BS = 256;

  // ---- layer 1 ----
  k_f32_to_bf16<<<blocksFor((size_t)NN * 128, BS), BS, 0, stream>>>(x, xbf, (size_t)NN * 128);
  k_transpose_bf16<<<blocksFor((size_t)128 * 256, BS), BS, 0, stream>>>(W1, w1t, 128, 256);
  {
    size_t waves = (size_t)(NN / 16) * (256 / 16);
    k_gemm_bf16<<<blocksFor(waves * 32, BS), BS, 0, stream>>>(xbf, w1t, h1, nullptr,
                                                              NN, 256, 128, 0);
  }
  k_att_reduce<<<blocksFor((size_t)NN * 2 * 32, BS), BS, 0, stream>>>(h1, as1, ad1, a1s, a1d, 2, 128);
  k_fill<<<blocksFor((size_t)NN * 2, BS), BS, 0, stream>>>(m1, -INFINITY, (size_t)NN * 2);
  k_fill<<<blocksFor((size_t)NN * 2, BS), BS, 0, stream>>>(z1, 0.0f, (size_t)NN * 2);
  k_fill<<<blocksFor((size_t)NN * 256, BS), BS, 0, stream>>>(agg1, 0.0f, (size_t)NN * 256);
  k_edge_max<<<blocksFor((size_t)EDGES * 2, BS), BS, 0, stream>>>(srcI, dstI, a1s, a1d, m1, 2);
  k_edge_exp<<<blocksFor((size_t)EDGES * 2, BS), BS, 0, stream>>>(srcI, dstI, a1s, a1d, m1, z1, ew1, 2);
  k_aggregate<<<blocksFor((size_t)EDGES * 2 * 32, BS), BS, 0, stream>>>(srcI, dstI, ew1, z1, h1, agg1, 2, 128);
  k_elu_bias_bf16<<<blocksFor((size_t)NN * 256, BS), BS, 0, stream>>>(agg1, b1, h1b, 256, (size_t)NN * 256);

  // ---- layer 2 ----
  k_transpose_bf16<<<blocksFor((size_t)256 * 128, BS), BS, 0, stream>>>(W2, w2t, 256, 128);
  {
    size_t waves = (size_t)(NN / 16) * (128 / 16);
    k_gemm_bf16<<<blocksFor(waves * 32, BS), BS, 0, stream>>>(h1b, w2t, h2, nullptr,
                                                              NN, 128, 256, 0);
  }
  k_att_reduce<<<blocksFor((size_t)NN * 32, BS), BS, 0, stream>>>(h2, as2, ad2, a2s, a2d, 1, 128);
  k_fill<<<blocksFor((size_t)NN, BS), BS, 0, stream>>>(m2, -INFINITY, (size_t)NN);
  k_fill<<<blocksFor((size_t)NN, BS), BS, 0, stream>>>(z2, 0.0f, (size_t)NN);
  k_fill<<<blocksFor((size_t)NN * 128, BS), BS, 0, stream>>>(agg2, 0.0f, (size_t)NN * 128);
  k_edge_max<<<blocksFor((size_t)EDGES, BS), BS, 0, stream>>>(srcI, dstI, a2s, a2d, m2, 1);
  k_edge_exp<<<blocksFor((size_t)EDGES, BS), BS, 0, stream>>>(srcI, dstI, a2s, a2d, m2, z2, ew2, 1);
  k_aggregate<<<blocksFor((size_t)EDGES * 32, BS), BS, 0, stream>>>(srcI, dstI, ew2, z2, h2, agg2, 1, 128);
  k_elu_bias_bf16<<<blocksFor((size_t)NN * 128, BS), BS, 0, stream>>>(agg2, b2, h2b, 128, (size_t)NN * 128);

  // ---- MLP head ----
  k_transpose_bf16<<<blocksFor((size_t)128 * 64, BS), BS, 0, stream>>>(fc1w, fc1t, 128, 64);
  {
    size_t waves = (size_t)(NN / 16) * (64 / 16);
    k_gemm_bf16<<<blocksFor(waves * 32, BS), BS, 0, stream>>>(h2b, fc1t, h3, fc1b,
                                                              NN, 64, 128, 1);
  }
  k_head<<<blocksFor((size_t)NN * 32, BS), BS, 0, stream>>>(h3, fc2w, fc2b, out, 64);
}
